// CategoryConditionedTextureDiffusion_24592982736957
// MI455X (gfx1250) — matrix operand order repaired and checked by reading, not compiled, not run on hardware
//
#include <hip/hip_runtime.h>

typedef __attribute__((ext_vector_type(16))) _Float16 v16h;
typedef __attribute__((ext_vector_type(8)))  float    v8f;

constexpr int Bn   = 4;
constexpr int Cc   = 24;
constexpr int NCLS = 6;
constexpr int TAPS = 49;      // 7x7
constexpr int OCW  = Cc * TAPS; // 1176
#define HW 16384              // 128*128

__device__ __forceinline__ _Float16 f2h(float x) { return (_Float16)x; }

// ---------------------------------------------------------------------------
// texture_encoder conv1: 3 -> 64, 3x3, zero pad, ReLU (tiny K, plain VALU)
// ---------------------------------------------------------------------------
__global__ __launch_bounds__(256)
void k_te1(const float* __restrict__ tf, const float* __restrict__ w,
           const float* __restrict__ bias, float* __restrict__ out)
{
    int id = blockIdx.x * blockDim.x + threadIdx.x;   // (b*64+oc)*HW + px
    if (id >= Bn * 64 * HW) return;
    int px = id & (HW - 1);
    int oc = (id >> 14) & 63;
    int b  = id >> 20;
    int h = px >> 7, x = px & 127;
    float s = bias[oc];
#pragma unroll
    for (int ic = 0; ic < 3; ++ic)
#pragma unroll
        for (int t = 0; t < 9; ++t) {
            int hh = h + t / 3 - 1, xx = x + t % 3 - 1;
            float v = (hh >= 0 && hh < 128 && xx >= 0 && xx < 128)
                        ? tf[((size_t)(b * 3 + ic) * HW) + hh * 128 + xx] : 0.f;
            s += v * w[(oc * 3 + ic) * 9 + t];
        }
    out[id] = fmaxf(s, 0.f);
}

// ---------------------------------------------------------------------------
// Generic 3x3 conv (zero pad) as implicit GEMM on v_wmma_f32_16x16x32_f16.
// Block = 128 threads = 4 waves. Each wave: 16 OC x 16 px f32 tile.
// Block tile: 16 OC x 64 px (one output row segment).
// K consumed 32 input channels at a time, one WMMA per 3x3 tap.
// EMB: channels >=128 come from class embedding (broadcast, zero at pad).
// ---------------------------------------------------------------------------
template<int IC, int OC, bool EMB, bool RELU>
__global__ __launch_bounds__(128)
void k_conv3(const float* __restrict__ in, const float* __restrict__ w,
             const float* __restrict__ bias, float* __restrict__ out,
             const float* __restrict__ emb, const int* __restrict__ labels)
{
    constexpr int ICM = EMB ? 128 : IC;               // channels held in `in`
    __shared__ __align__(32) _Float16 ilds[3][68][32]; // [row][col][ch] ch innermost
    __shared__ __align__(32) _Float16 wlds[9][32][16]; // A-fragment order per tap

    const int b   = blockIdx.z;
    const int ocb = blockIdx.y * 16;
    const int oh  = blockIdx.x >> 1;
    const int pxb = (blockIdx.x & 1) * 64;
    const int tid = threadIdx.x;
    const int wv  = tid >> 5, ln = tid & 31;
    const int lab = EMB ? labels[b] : 0;

    v8f acc = {};
    for (int icb = 0; icb < IC; icb += 32) {
        // stage activations: 3 rows x 66 cols x 32 ch, f32 -> f16
        for (int i = tid; i < 3 * 66 * 32; i += 128) {
            int ch = i & 31, cc = (i >> 5) % 66, r = (i >> 5) / 66;
            int gh = oh + r - 1, gx = pxb + cc - 1, gic = icb + ch;
            float v = 0.f;
            if (gh >= 0 && gh < 128 && gx >= 0 && gx < 128) {
                if (EMB && gic >= 128) v = emb[lab * 128 + (gic - 128)];
                else v = in[((size_t)(b * ICM + gic) * HW) + gh * 128 + gx];
            }
            ilds[r][cc][ch] = f2h(v);
        }
        // stage weights pre-swizzled into WMMA A-fragment layout
        for (int i = tid; i < 9 * 32 * 16; i += 128) {
            int e = i & 15, lane = (i >> 4) & 31, tap = i >> 9;
            int p = e >> 1, wi = e & 1;
            int icl = ((p & 3) * 2 + wi) + ((lane & 16) ? 16 : 0) + ((p >= 4) ? 8 : 0);
            int M = lane & 15;
            wlds[tap][lane][e] =
                f2h(w[((size_t)(ocb + M) * IC + icb + icl) * 9 + tap]);
        }
        __syncthreads();
#pragma unroll
        for (int tap = 0; tap < 9; ++tap) {
            const int ki = tap / 3, kj = tap % 3;
            v16h a = *(const v16h*)(&wlds[tap][ln][0]);            // 2x ds_read_b128
            const int cl   = wv * 16 + (ln & 15) + kj;             // N-pixel column
            const int koff = (ln & 16) ? 16 : 0;
            v16h bf = *(const v16h*)(&ilds[ki][cl][koff]);         // 2x ds_read_b128
            acc = __builtin_amdgcn_wmma_f32_16x16x32_f16(
                      false, a, false, bf, (short)0, acc, false, false);
        }
        __syncthreads();
    }
    const int px = pxb + wv * 16 + (ln & 15);
#pragma unroll
    for (int m = 0; m < 8; ++m) {
        int oc = ocb + m + ((ln & 16) ? 8 : 0);
        float v = acc[m] + bias[oc];
        if (RELU) v = fmaxf(v, 0.f);
        out[((size_t)(b * OC + oc) * HW) + oh * 128 + px] = v;
    }
}

// ---------------------------------------------------------------------------
// fusion conv2: 1x1, 256 -> 1176  == GEMM [1176 x 256] x [256 x HW] per batch
// Block: 16 OC x 64 px; full K (256) staged once; 8 WMMAs per wave.
// ---------------------------------------------------------------------------
__global__ __launch_bounds__(128)
void k_fuse2(const float* __restrict__ x, const float* __restrict__ w,
             const float* __restrict__ bias, float* __restrict__ out)
{
    __shared__ __align__(32) _Float16 xl[64][256];      // [px][k] 32 KB
    __shared__ __align__(32) _Float16 wl[8][32][16];    // A-fragment order, 8 KB
    const int b   = blockIdx.z;
    const int ocb = blockIdx.y * 16;
    const int pxb = blockIdx.x * 64;
    const int tid = threadIdx.x, wv = tid >> 5, ln = tid & 31;

    for (int i = tid; i < 64 * 256; i += 128) {
        int k = i & 255, p = i >> 8;
        xl[p][k] = f2h(x[((size_t)(b * 256 + k) * HW) + pxb + p]);
    }
    for (int i = tid; i < 8 * 32 * 16; i += 128) {
        int e = i & 15, lane = (i >> 4) & 31, kb = i >> 9;
        int p = e >> 1, wi = e & 1;
        int icl = ((p & 3) * 2 + wi) + ((lane & 16) ? 16 : 0) + ((p >= 4) ? 8 : 0);
        int oc = ocb + (lane & 15);
        int k  = kb * 32 + icl;
        wl[kb][lane][e] = f2h(oc < OCW ? w[oc * 256 + k] : 0.f);
    }
    __syncthreads();

    v8f acc = {};
    const int pl   = wv * 16 + (ln & 15);
    const int koff = (ln & 16) ? 16 : 0;
#pragma unroll
    for (int kb = 0; kb < 8; ++kb) {
        v16h a  = *(const v16h*)(&wl[kb][ln][0]);
        v16h bf = *(const v16h*)(&xl[pl][kb * 32 + koff]);
        acc = __builtin_amdgcn_wmma_f32_16x16x32_f16(
                  false, a, false, bf, (short)0, acc, false, false);
    }
    const int px = pxb + pl;
#pragma unroll
    for (int m = 0; m < 8; ++m) {
        int oc = ocb + m + ((ln & 16) ? 8 : 0);
        if (oc < OCW)
            out[((size_t)(b * OCW + oc) * HW) + px] = acc[m] + bias[oc];
    }
}

// ---------------------------------------------------------------------------
// class_gate conv2 (64->6, 1x1) + sigmoid + one-hot mask + intensity/cond maps
// ---------------------------------------------------------------------------
__global__ __launch_bounds__(256)
void k_gate2(const float* __restrict__ g1, const float* __restrict__ w,
             const float* __restrict__ bias, const int* __restrict__ labels,
             const float* __restrict__ ci,
             float* __restrict__ out_cg, float* __restrict__ inten,
             float* __restrict__ condmap)
{
    int id = blockIdx.x * blockDim.x + threadIdx.x;   // b*HW + px
    if (id >= Bn * HW) return;
    int px = id & (HW - 1), b = id >> 14;
    float a[NCLS];
#pragma unroll
    for (int k = 0; k < NCLS; ++k) a[k] = bias[k];
    for (int ch = 0; ch < 64; ++ch) {
        float g = g1[((size_t)(b * 64 + ch) * HW) + px];
#pragma unroll
        for (int k = 0; k < NCLS; ++k) a[k] += g * w[k * 64 + ch];
    }
    int lab = labels[b];
    float sl = 0.f;
#pragma unroll
    for (int k = 0; k < NCLS; ++k) {
        float s = 1.f / (1.f + __expf(-a[k]));
        float cg = (k == lab) ? s : 0.f;
        out_cg[((size_t)(b * NCLS + k) * HW) + px] = cg;
        if (k == lab) sl = s;
    }
    float it = sl * ci[lab];            // intensity_map = sum_k gate_k*onehot*ci
    inten[id] = it;
    condmap[((size_t)b * HW) + px] = it * sl;  // intensity * class_gate.sum
}

// ---------------------------------------------------------------------------
// weights = softmax over 49 taps of (raw * intensity); store f16 (halves the
// diffusion-loop traffic: 154 MB fits mostly in the 192 MB L2)
// ---------------------------------------------------------------------------
__global__ __launch_bounds__(256)
void k_softmax(const float* __restrict__ wraw, const float* __restrict__ inten,
               _Float16* __restrict__ wsoft)
{
    int id = blockIdx.x * blockDim.x + threadIdx.x;  // (b*C + c)*HW + px
    if (id >= Bn * Cc * HW) return;
    int px = id & (HW - 1);
    int q = id >> 14, c = q % Cc, b = q / Cc;
    float it = inten[b * HW + px];
    const float* base = wraw + ((size_t)(b * OCW + c * TAPS) * HW) + px;
    float v[TAPS];
    float mx = -1e30f;
#pragma unroll
    for (int t = 0; t < TAPS; ++t) { v[t] = base[(size_t)t * HW] * it; mx = fmaxf(mx, v[t]); }
    float s = 0.f;
#pragma unroll
    for (int t = 0; t < TAPS; ++t) { v[t] = __expf(v[t] - mx); s += v[t]; }
    float inv = 1.f / s;
    _Float16* ob = wsoft + ((size_t)(b * OCW + c * TAPS) * HW) + px;
#pragma unroll
    for (int t = 0; t < TAPS; ++t) ob[(size_t)t * HW] = f2h(v[t] * inv);
}

// ---------------------------------------------------------------------------
// one diffusion step: 7x7 per-pixel weighted sum, edge-clamped latent
// ---------------------------------------------------------------------------
__global__ __launch_bounds__(256)
void k_diffuse(const float* __restrict__ lin, const _Float16* __restrict__ wsoft,
               float* __restrict__ lout)
{
    int id = blockIdx.x * blockDim.x + threadIdx.x;  // (b*C + c)*HW + px
    if (id >= Bn * Cc * HW) return;
    int px = id & (HW - 1);
    int q = id >> 14, c = q % Cc, b = q / Cc;
    int h = px >> 7, x = px & 127;
    const _Float16* wb = wsoft + ((size_t)(b * OCW + c * TAPS) * HW) + px;
    const float* lb = lin + ((size_t)(b * Cc + c) * HW);
    __builtin_prefetch(wb, 0, 0);
    float s = 0.f;
#pragma unroll
    for (int t = 0; t < TAPS; ++t) {
        int hh = h + t / 7 - 3, xx = x + t % 7 - 3;
        hh = min(max(hh, 0), 127);
        xx = min(max(xx, 0), 127);
        s += (float)wb[(size_t)t * HW] * lb[hh * 128 + xx];
    }
    lout[id] = s;
}

// ---------------------------------------------------------------------------
extern "C" void kernel_launch(void* const* d_in, const int* in_sizes, int n_in,
                              void* d_out, int out_size, void* d_ws, size_t ws_size,
                              hipStream_t stream)
{
    const float* depth = (const float*)d_in[0];
    const float* tf    = (const float*)d_in[1];
    const int*   lab   = (const int*)  d_in[2];
    const float* w_te1 = (const float*)d_in[3];
    const float* b_te1 = (const float*)d_in[4];
    const float* w_te2 = (const float*)d_in[5];
    const float* b_te2 = (const float*)d_in[6];
    const float* emb   = (const float*)d_in[7];
    const float* w_f1  = (const float*)d_in[8];
    const float* b_f1  = (const float*)d_in[9];
    const float* w_f2  = (const float*)d_in[10];
    const float* b_f2  = (const float*)d_in[11];
    const float* w_g1  = (const float*)d_in[12];
    const float* b_g1  = (const float*)d_in[13];
    const float* w_g2  = (const float*)d_in[14];
    const float* b_g2  = (const float*)d_in[15];
    const float* ci    = (const float*)d_in[16];

    float* out_enh = (float*)d_out;                        // [4,24,128,128]
    float* out_cg  = out_enh + (size_t)Bn * Cc * HW;       // [4,6,128,128]
    float* out_cm  = out_cg  + (size_t)Bn * NCLS * HW;     // [4,1,128,128]

    float* ws = (float*)d_ws;
    size_t o = 0;
    float* tex1 = ws + o; o += (size_t)Bn * 64  * HW;
    float* tex2 = ws + o; o += (size_t)Bn * 128 * HW;
    float* g1   = ws + o; o += (size_t)Bn * 64  * HW;
    float* intn = ws + o; o += (size_t)Bn * HW;
    float* f1   = ws + o; o += (size_t)Bn * 256 * HW;
    float* wraw = ws + o; o += (size_t)Bn * OCW * HW;
    float* latA = ws + o; o += (size_t)Bn * Cc  * HW;
    float* latB = ws + o; o += (size_t)Bn * Cc  * HW;
    _Float16* wsoft = (_Float16*)(ws + o);                 // [4,1176,HW] f16

    // texture encoder
    k_te1<<<(Bn * 64 * HW) / 256, 256, 0, stream>>>(tf, w_te1, b_te1, tex1);
    k_conv3<64, 128, false, true><<<dim3(256, 8, Bn), 128, 0, stream>>>(
        tex1, w_te2, b_te2, tex2, nullptr, nullptr);
    // class gate
    k_conv3<128, 64, false, true><<<dim3(256, 4, Bn), 128, 0, stream>>>(
        tex2, w_g1, b_g1, g1, nullptr, nullptr);
    k_gate2<<<(Bn * HW) / 256, 256, 0, stream>>>(
        g1, w_g2, b_g2, lab, ci, out_cg, intn, out_cm);
    // fusion (second 128 input channels = broadcast class embedding)
    k_conv3<256, 256, true, true><<<dim3(256, 16, Bn), 128, 0, stream>>>(
        tex2, w_f1, b_f1, f1, emb, lab);
    k_fuse2<<<dim3(HW / 64, 74, Bn), 128, 0, stream>>>(f1, w_f2, b_f2, wraw);
    // per-pixel softmax over 49 taps, scaled by intensity, stored f16
    k_softmax<<<(Bn * Cc * HW) / 256, 256, 0, stream>>>(wraw, intn, wsoft);
    // 8 diffusion steps, ping-pong; final step writes straight to d_out
    const float* cur = depth;
    for (int s = 0; s < 8; ++s) {
        float* nxt = (s == 7) ? out_enh : ((s & 1) ? latB : latA);
        k_diffuse<<<(Bn * Cc * HW) / 256, 256, 0, stream>>>(cur, wsoft, nxt);
        cur = nxt;
    }
}
